// Transformer_38208029065922
// MI455X (gfx1250) — compile-verified, matrix operand order checked by reading, not run on hardware
//
#include <hip/hip_runtime.h>

#define B_SZ    8
#define T_SEQ   2048
#define D_MODEL 256
#define N_HEAD  4
#define HSZ     64
#define DFF_SZ  1024
#define N_LAYER 4

typedef __attribute__((ext_vector_type(16))) __bf16          v16bf;
typedef __attribute__((ext_vector_type(8)))  float           v8f;
typedef __attribute__((ext_vector_type(16))) unsigned short  us16;
typedef __attribute__((ext_vector_type(8)))  unsigned short  us8;

__device__ __forceinline__ unsigned short f2bf(float f) {
    unsigned int u = __builtin_bit_cast(unsigned int, f);
    u += 0x7FFFu + ((u >> 16) & 1u);          // round-to-nearest-even
    return (unsigned short)(u >> 16);
}
__device__ __forceinline__ float bf2f(unsigned short h) {
    unsigned int u = ((unsigned int)h) << 16;
    return __builtin_bit_cast(float, u);
}

__device__ __forceinline__ v8f vzero8() {
    v8f z = {0.f, 0.f, 0.f, 0.f, 0.f, 0.f, 0.f, 0.f};
    return z;
}

// 32-bit wave-relative LDS byte address of a __shared__ object (AS3 ptrtoint),
// as required for VDST of GLOBAL_LOAD_ASYNC_TO_LDS (ISA 08 §4.4).
__device__ __forceinline__ unsigned lds_addr(const unsigned short* p) {
    return (unsigned)(unsigned long long)
        (__attribute__((address_space(3))) const unsigned short*)p;
}

// Async global -> LDS 128-bit copy, per-lane, tracked by ASYNCcnt.
__device__ __forceinline__ void async_g2l_b128(unsigned lds_off, const void* gptr) {
    unsigned long long ga = (unsigned long long)gptr;
    asm volatile("global_load_async_to_lds_b128 %0, %1, off"
                 :: "v"(lds_off), "v"(ga) : "memory");
}
__device__ __forceinline__ void wait_async0() {
    asm volatile("s_wait_asynccnt 0x0" ::: "memory");
}

// Assemble a 16x32 bf16 WMMA fragment (A-layout; B uses the same layout on
// transposed [n][k] storage). Per ISA 7.12.2: lane holds major index lane&15,
// K elements (lane>>4)*8 + {0..7} and +{16..23}.
__device__ __forceinline__ v16bf load_frag(const unsigned short* base, int stride, int major0) {
    int lane = threadIdx.x & 31;
    const unsigned short* p = base + (size_t)(major0 + (lane & 15)) * stride + ((lane >> 4) << 3);
    us8 lo = *(const us8*)p;
    us8 hi = *(const us8*)(p + 16);
    us16 u = __builtin_shufflevector(lo, hi, 0,1,2,3,4,5,6,7,8,9,10,11,12,13,14,15);
    return __builtin_bit_cast(v16bf, u);
}

__device__ __forceinline__ v8f wmma32(v16bf a, v16bf b, v8f c) {
    return __builtin_amdgcn_wmma_f32_16x16x32_bf16(false, a, false, b, (short)0, c, false, false);
}

// ---------------- embedding: x = tok_emb[idx] + pos_emb ----------------
__global__ void __launch_bounds__(256) embed_kernel(const int* __restrict__ idx,
                                                    const float* __restrict__ tok_emb,
                                                    const float* __restrict__ pos_emb,
                                                    float* __restrict__ x) {
    int row = blockIdx.x;            // b*T + t
    int t   = row & (T_SEQ - 1);
    int d   = threadIdx.x;
    x[(size_t)row * D_MODEL + d] =
        tok_emb[(size_t)idx[row] * D_MODEL + d] + pos_emb[(size_t)t * D_MODEL + d];
}

// ---------------- layernorm over D=256, bf16 (and optional f32) out ----
__global__ void __launch_bounds__(256) ln_kernel(const float* __restrict__ x,
                                                 const float* __restrict__ g,
                                                 const float* __restrict__ b,
                                                 unsigned short* __restrict__ outb,
                                                 float* __restrict__ outf,
                                                 int write_f32) {
    __shared__ float red[256];
    int row = blockIdx.x, tid = threadIdx.x;
    float v = x[(size_t)row * D_MODEL + tid];
    red[tid] = v;
    __syncthreads();
    for (int s = 128; s > 0; s >>= 1) {
        if (tid < s) red[tid] += red[tid + s];
        __syncthreads();
    }
    float mu = red[0] * (1.f / D_MODEL);
    __syncthreads();
    float d = v - mu;
    red[tid] = d * d;
    __syncthreads();
    for (int s = 128; s > 0; s >>= 1) {
        if (tid < s) red[tid] += red[tid + s];
        __syncthreads();
    }
    float var = red[0] * (1.f / D_MODEL);
    float y = d * rsqrtf(var + 1e-5f) * g[tid] + b[tid];
    outb[(size_t)row * D_MODEL + tid] = f2bf(y);
    if (write_f32) outf[(size_t)row * D_MODEL + tid] = y;
}

// ---------------- generic WMMA GEMM: C = A(bf16 MxK) * W(f32 KxN) ------
// wmode: 0 = W[k*N+n] row-major, 1 = QKV head layout W[h][k][s], n=(h,s)
// omode: 0 = bf16 out row-major, 1 = bf16 out as [B,H,T,HS], 2 = resid f32 +=
__global__ void __launch_bounds__(256) gemm_bf16_kernel(const unsigned short* __restrict__ A,
                                                        const float* __restrict__ W,
                                                        const float* __restrict__ bias,
                                                        unsigned short* __restrict__ outb,
                                                        float* __restrict__ resid,
                                                        int M, int N, int K,
                                                        int wmode, int omode, int relu) {
    __shared__ __align__(16) unsigned short As[128 * 32];
    __shared__ __align__(16) unsigned short Ws[64 * 32];   // transposed [n][k]
    int tid = threadIdx.x, lane = tid & 31, wave = tid >> 5;
    int wm = wave >> 1, wn = wave & 1;                     // 4 x 2 wave grid
    int mblk = blockIdx.y * 128, nblk = blockIdx.x * 64;

    v8f acc[2][2];
    for (int i = 0; i < 2; i++)
        for (int j = 0; j < 2; j++) acc[i][j] = vzero8();

    for (int k0 = 0; k0 < K; k0 += 32) {
        __syncthreads();
        // stage A tile 128x32: async global->LDS (ASYNCcnt), no VGPR bounce
        #pragma unroll
        for (int i = 0; i < 2; i++) {
            int e = (tid + i * 256) * 8;
            int r = e >> 5, c = e & 31;
            async_g2l_b128(lds_addr(As + r * 32 + c),
                           A + (size_t)(mblk + r) * K + k0 + c);
        }
        if (k0 + 32 < K)
            __builtin_prefetch(A + (size_t)(mblk + (tid >> 1)) * K + k0 + 32, 0, 1);
        // stage W tile 32x64, f32 -> bf16, store transposed [n][k]
        // (overlaps with the async A copy above)
        #pragma unroll
        for (int j = 0; j < 8; j++) {
            int e = tid * 8 + j;
            int kk = e >> 6, nn = e & 63;
            int ng = nblk + nn, kg = k0 + kk;
            float wv = (wmode == 0)
                           ? W[(size_t)kg * N + ng]
                           : W[((size_t)(ng >> 6) * K + kg) * HSZ + (ng & 63)];
            Ws[nn * 32 + kk] = f2bf(wv);
        }
        wait_async0();
        __syncthreads();
        v16bf a0 = load_frag(As, 32, wm * 32);
        v16bf a1 = load_frag(As, 32, wm * 32 + 16);
        v16bf b0 = load_frag(Ws, 32, wn * 32);
        v16bf b1 = load_frag(Ws, 32, wn * 32 + 16);
        acc[0][0] = wmma32(a0, b0, acc[0][0]);
        acc[0][1] = wmma32(a0, b1, acc[0][1]);
        acc[1][0] = wmma32(a1, b0, acc[1][0]);
        acc[1][1] = wmma32(a1, b1, acc[1][1]);
    }

    // epilogue (C layout: lane holds col lane&15, rows r + 8*(lane>>4))
    #pragma unroll
    for (int mi = 0; mi < 2; mi++)
        #pragma unroll
        for (int ni = 0; ni < 2; ni++) {
            int n = nblk + wn * 32 + ni * 16 + (lane & 15);
            float bv = bias ? bias[n] : 0.f;
            #pragma unroll
            for (int r = 0; r < 8; r++) {
                int m = mblk + wm * 32 + mi * 16 + r + ((lane >> 4) << 3);
                float val = acc[mi][ni][r] + bv;
                if (relu) val = fmaxf(val, 0.f);
                if (omode == 0) {
                    outb[(size_t)m * N + n] = f2bf(val);
                } else if (omode == 1) {
                    int bb = m >> 11, tt = m & (T_SEQ - 1);
                    int hh = n >> 6, ss = n & 63;
                    outb[(((size_t)(bb * N_HEAD + hh)) * T_SEQ + tt) * HSZ + ss] = f2bf(val);
                } else {
                    resid[(size_t)m * N + n] += val;
                }
            }
        }
}

// ---------------- flash attention, bf16 WMMA, f32 online softmax -------
__global__ void __launch_bounds__(256) attn_kernel(const unsigned short* __restrict__ qg,
                                                   const unsigned short* __restrict__ kg,
                                                   const unsigned short* __restrict__ vg,
                                                   unsigned short* __restrict__ og) {
    __shared__ __align__(16) unsigned short Ks[32 * 64];       // [key][s]
    __shared__ __align__(16) unsigned short Vt[64 * 32];       // [s][key]
    __shared__ __align__(16) unsigned short Ps[8 * 16 * 32];   // per-wave P
    int tid = threadIdx.x, lane = tid & 31, wave = tid >> 5;
    int half = lane >> 4;
    int qblk = blockIdx.x, bh = blockIdx.y;
    int b = bh >> 2, h = bh & 3;
    const unsigned short* Q  = qg + (size_t)bh * T_SEQ * HSZ;
    const unsigned short* Kp = kg + (size_t)bh * T_SEQ * HSZ;
    const unsigned short* Vp = vg + (size_t)bh * T_SEQ * HSZ;
    int q0 = qblk * 128;

    // Q fragments (16 rows x 64) held in registers for the whole pass
    v16bf qa0 = load_frag(Q,      HSZ, q0 + wave * 16);
    v16bf qa1 = load_frag(Q + 32, HSZ, q0 + wave * 16);

    float mrow[8], lrow[8];
    v8f accv[4];
    #pragma unroll
    for (int r = 0; r < 8; r++) { mrow[r] = -1e30f; lrow[r] = 0.f; }
    #pragma unroll
    for (int t = 0; t < 4; t++) accv[t] = vzero8();

    const float scale = 0.0625f;   // D_MODEL^-0.5 (reference scales by n_embd)
    unsigned short* Pw = Ps + wave * (16 * 32);
    int jend = q0 + 128;

    for (int j0 = 0; j0 < jend; j0 += 32) {
        __syncthreads();
        {   // cooperative K/V stage: 32 keys x 64
            int e = tid * 8;
            int key = e >> 6, s = e & 63;
            // K: async global->LDS, overlapped with the V transpose below
            async_g2l_b128(lds_addr(Ks + key * 64 + s),
                           Kp + (size_t)(j0 + key) * HSZ + s);
            // V: through VGPRs (needs in-register transpose)
            us8 vv = *(const us8*)(Vp + (size_t)(j0 + key) * HSZ + s);
            #pragma unroll
            for (int qi = 0; qi < 8; qi++) Vt[(s + qi) * 32 + key] = vv[qi];
            wait_async0();
        }
        __syncthreads();

        // S = Q * K^T  (two 16x16 key tiles, K-dim = HS = 64)
        v8f s0 = vzero8(), s1 = vzero8();
        s0 = wmma32(qa0, load_frag(Ks,      64, 0),  s0);
        s0 = wmma32(qa1, load_frag(Ks + 32, 64, 0),  s0);
        s1 = wmma32(qa0, load_frag(Ks,      64, 16), s1);
        s1 = wmma32(qa1, load_frag(Ks + 32, 64, 16), s1);

        int key0 = j0 + (lane & 15);
        #pragma unroll
        for (int r = 0; r < 8; r++) {
            int qrow = q0 + wave * 16 + r + half * 8;
            float f0 = s0[r] * scale; if (key0      > qrow) f0 = -1e30f;
            float f1 = s1[r] * scale; if (key0 + 16 > qrow) f1 = -1e30f;
            float rm = fmaxf(f0, f1);
            #pragma unroll
            for (int msk = 8; msk >= 1; msk >>= 1) rm = fmaxf(rm, __shfl_xor(rm, msk, 16));
            float mnew  = fmaxf(mrow[r], rm);
            float alpha = __expf(mrow[r] - mnew);
            float p0 = __expf(f0 - mnew);
            float p1 = __expf(f1 - mnew);
            float rs = p0 + p1;
            #pragma unroll
            for (int msk = 8; msk >= 1; msk >>= 1) rs += __shfl_xor(rs, msk, 16);
            lrow[r] = lrow[r] * alpha + rs;
            mrow[r] = mnew;
            #pragma unroll
            for (int t = 0; t < 4; t++) accv[t][r] *= alpha;
            Pw[(r + half * 8) * 32 + (lane & 15)]      = f2bf(p0);
            Pw[(r + half * 8) * 32 + 16 + (lane & 15)] = f2bf(p1);
        }
        __syncthreads();   // P C-layout -> A-layout via LDS

        v16bf pa = load_frag(Pw, 32, 0);
        #pragma unroll
        for (int t = 0; t < 4; t++)
            accv[t] = wmma32(pa, load_frag(Vt, 32, t * 16), accv[t]);
    }

    // normalize and write o as [B,T,D] (heads concatenated)
    #pragma unroll
    for (int r = 0; r < 8; r++) {
        int row = q0 + wave * 16 + r + half * 8;
        float inv = 1.f / lrow[r];
        #pragma unroll
        for (int t = 0; t < 4; t++) {
            int n = t * 16 + (lane & 15);
            og[((size_t)(b * T_SEQ + row)) * D_MODEL + h * HSZ + n] = f2bf(accv[t][r] * inv);
        }
    }
}

// ---------------- mean-pool + classifier head --------------------------
__global__ void __launch_bounds__(256) head_kernel(const float* __restrict__ xf,
                                                   const float* __restrict__ Wpre,
                                                   const float* __restrict__ bpre,
                                                   const float* __restrict__ Wcls,
                                                   const float* __restrict__ bcls,
                                                   float* __restrict__ out) {
    __shared__ float pool[256];
    __shared__ float y[256];
    int b = blockIdx.x, tid = threadIdx.x;
    float s = 0.f;
    for (int t = 0; t < T_SEQ; t++)
        s += xf[((size_t)b * T_SEQ + t) * D_MODEL + tid];
    pool[tid] = s * (1.f / T_SEQ);
    __syncthreads();
    float acc = bpre[tid];
    for (int d = 0; d < D_MODEL; d++) acc += pool[d] * Wpre[(size_t)d * D_MODEL + tid];
    y[tid] = acc;
    __syncthreads();
    if (tid < 8) {
        float a = bcls[tid];
        for (int e = 0; e < D_MODEL; e++) a += y[e] * Wcls[(size_t)e * 8 + tid];
        out[b * 8 + tid] = a;
    }
}

extern "C" void kernel_launch(void* const* d_in, const int* in_sizes, int n_in,
                              void* d_out, int out_size, void* d_ws, size_t ws_size,
                              hipStream_t stream) {
    (void)in_sizes; (void)n_in; (void)out_size; (void)ws_size;
    const int*   idx     = (const int*)d_in[0];
    const float* tok_emb = (const float*)d_in[1];
    const float* pos_emb = (const float*)d_in[2];
    const float* Wq      = (const float*)d_in[3];
    const float* Wk      = (const float*)d_in[4];
    const float* Wv      = (const float*)d_in[5];
    const float* Wproj   = (const float*)d_in[6];
    const float* bproj   = (const float*)d_in[7];
    const float* ln1_g   = (const float*)d_in[8];
    const float* ln1_b   = (const float*)d_in[9];
    const float* ln2_g   = (const float*)d_in[10];
    const float* ln2_b   = (const float*)d_in[11];
    const float* W1      = (const float*)d_in[12];
    const float* b1      = (const float*)d_in[13];
    const float* W2      = (const float*)d_in[14];
    const float* b2      = (const float*)d_in[15];
    const float* lnf_g   = (const float*)d_in[16];
    const float* lnf_b   = (const float*)d_in[17];
    const float* Wpre    = (const float*)d_in[18];
    const float* bpre    = (const float*)d_in[19];
    const float* Wcls    = (const float*)d_in[20];
    const float* bcls    = (const float*)d_in[21];

    const int rows = B_SZ * T_SEQ;                         // 16384
    char* ws = (char*)d_ws;
    size_t off = 0;
    float*          x  = (float*)(ws + off);          off += (size_t)rows * D_MODEL * 4;
    unsigned short* hb = (unsigned short*)(ws + off); off += (size_t)rows * D_MODEL * 2;
    unsigned short* qb = (unsigned short*)(ws + off); off += (size_t)rows * D_MODEL * 2;
    unsigned short* kb = (unsigned short*)(ws + off); off += (size_t)rows * D_MODEL * 2;
    unsigned short* vb = (unsigned short*)(ws + off); off += (size_t)rows * D_MODEL * 2;
    unsigned short* ob = (unsigned short*)(ws + off); off += (size_t)rows * D_MODEL * 2;
    unsigned short* mb = (unsigned short*)(ws + off); off += (size_t)rows * DFF_SZ * 2;
    float* xf = (float*)qb;   // reuse q+k region (16 MB) after attention is done

    embed_kernel<<<rows, 256, 0, stream>>>(idx, tok_emb, pos_emb, x);

    const dim3 g256(D_MODEL / 64, rows / 128);    // (4,128)
    const dim3 g1024(DFF_SZ / 64, rows / 128);    // (16,128)

    for (int l = 0; l < N_LAYER; l++) {
        ln_kernel<<<rows, 256, 0, stream>>>(x, ln1_g + l * D_MODEL, ln1_b + l * D_MODEL,
                                            hb, (float*)nullptr, 0);
        gemm_bf16_kernel<<<g256, 256, 0, stream>>>(hb, Wq + (size_t)l * 65536, nullptr,
                                                   qb, nullptr, rows, D_MODEL, D_MODEL, 1, 1, 0);
        gemm_bf16_kernel<<<g256, 256, 0, stream>>>(hb, Wk + (size_t)l * 65536, nullptr,
                                                   kb, nullptr, rows, D_MODEL, D_MODEL, 1, 1, 0);
        gemm_bf16_kernel<<<g256, 256, 0, stream>>>(hb, Wv + (size_t)l * 65536, nullptr,
                                                   vb, nullptr, rows, D_MODEL, D_MODEL, 1, 1, 0);
        attn_kernel<<<dim3(T_SEQ / 128, B_SZ * N_HEAD), 256, 0, stream>>>(qb, kb, vb, ob);
        gemm_bf16_kernel<<<g256, 256, 0, stream>>>(ob, Wproj + (size_t)l * 65536,
                                                   bproj + l * D_MODEL, nullptr, x,
                                                   rows, D_MODEL, D_MODEL, 0, 2, 0);
        ln_kernel<<<rows, 256, 0, stream>>>(x, ln2_g + l * D_MODEL, ln2_b + l * D_MODEL,
                                            hb, (float*)nullptr, 0);
        gemm_bf16_kernel<<<g1024, 256, 0, stream>>>(hb, W1 + (size_t)l * 262144,
                                                    b1 + l * DFF_SZ, mb, nullptr,
                                                    rows, DFF_SZ, D_MODEL, 0, 0, 1);
        gemm_bf16_kernel<<<g256, 256, 0, stream>>>(mb, W2 + (size_t)l * 262144,
                                                   b2 + l * D_MODEL, nullptr, x,
                                                   rows, D_MODEL, DFF_SZ, 0, 2, 0);
    }

    ln_kernel<<<rows, 256, 0, stream>>>(x, lnf_g, lnf_b, hb, xf, 1);
    head_kernel<<<B_SZ, 256, 0, stream>>>(xf, Wpre, bpre, Wcls, bcls, (float*)d_out);
}
